// Cfc_37546604101675
// MI455X (gfx1250) — compile-verified, matrix operand order checked by reading, not run on hardware
//
#include <hip/hip_runtime.h>
#include <math.h>

typedef __attribute__((ext_vector_type(16))) _Float16 v16h;
typedef __attribute__((ext_vector_type(8)))  float    v8f;

#define B_  512
#define S_  168
#define F_  12
#define H_  128
#define O_  3
#define U_  256
#define K0P 160   // layer0 K padded 140 -> 160
#define KT0 5
#define NT0 16
#define KT1 8
#define NT1 16
#define KTH 8
#define NTH 8
#define KTC 4

// LDS layout (bytes) for the RNN kernel (dynamic shared memory)
#define LDS_HB    0                       // float[3][16*128]  = 24576
#define LDS_ZIN   24576                   // f16[16*160]       = 5120
#define LDS_Z1    29696                   // f16[16*256]       = 8192
#define LDS_Z2    37888                   // f16[16*256]       = 8192
#define LDS_HW    46080                   // f16 head frags 3*65536 + wfc 4096
#define HW_BYTES  (NTH * KTH * 512 * 2)   // 65536 per head
#define LDS_TOTAL (46080 + 3 * HW_BYTES + KTC * 512 * 2)   // 246784

#define WMMA_F16(a, b, c) \
    __builtin_amdgcn_wmma_f32_16x16x32_f16(false, (a), false, (b), (short)0, (c), false, false)

// ---------------------------------------------------------------------------
// A-fragment (16x32 f16, M x K) gather from row-major LDS activation buffer.
// ISA layout: lane m=lane&15; half i: K = (i&7) + 16*(i>>3) + 8*(lane>>4).
// Consecutive half pairs are consecutive K -> 8 aligned 32-bit LDS loads.
// ---------------------------------------------------------------------------
__device__ __forceinline__ v16h load_afrag(const _Float16* base, int ld, int lane, int kbase)
{
    union { v16h v; unsigned u[8]; } r;
    const _Float16* p = base + (lane & 15) * ld + kbase + ((lane >> 4) << 3);
#pragma unroll
    for (int j = 0; j < 8; ++j) {
        int k = 2 * (j & 3) + ((j >> 2) << 4);
        r.u[j] = *(const unsigned*)(p + k);
    }
    return r.v;
}

// B fragment: pre-swizzled (global or LDS), one contiguous 32B load per lane.
__device__ __forceinline__ v16h load_bfrag(const _Float16* frags, int frag, int lane)
{
    return *(const v16h*)(frags + ((size_t)frag << 9) + (lane << 4));
}

__device__ __forceinline__ float silu_f(float x) { return x / (1.f + __expf(-x)); }
__device__ __forceinline__ float sigm_f(float x) { return 1.f / (1.f + __expf(-x)); }
// exact identity tanh(x) = 1 - 2/(exp(2x)+1); saturates correctly at +-inf
__device__ __forceinline__ float tanh_f(float x) { return 1.f - 2.f / (__expf(2.f * x) + 1.f); }

__device__ __forceinline__ void copy_g2l(_Float16* dst, const _Float16* src,
                                         int halfs, int tid, int nthreads)
{
    const uint4* s = (const uint4*)src;
    uint4* d = (uint4*)dst;
    int n = halfs >> 3;                 // 8 halfs per 16B
    for (int i = tid; i < n; i += nthreads) d[i] = s[i];
}

// ---------------------------------------------------------------------------
// Weight swizzle: f32 (Nout x Kin) row-major -> f16 WMMA B-fragments.
// frag = nt*KT + kt ; thread t: lane=t>>4, half i=t&15.
// K = kt*32 + (i&7) + 16*(i>>3) + 8*(lane>>4); N = nt*16 + (lane&15).
// Optional W2 added in (used to fold Wta+Wtb).
// ---------------------------------------------------------------------------
__global__ __launch_bounds__(512) void swizzle_kernel(
    const float* __restrict__ W, const float* __restrict__ W2,
    _Float16* __restrict__ dst, int Nout, int Kin, int KT)
{
    int frag = blockIdx.x;
    int kt = frag % KT, nt = frag / KT;
    int t = threadIdx.x;
    int lane = t >> 4, i = t & 15;
    int K = kt * 32 + (i & 7) + ((i >> 3) << 4) + ((lane >> 4) << 3);
    int N = nt * 16 + (lane & 15);
    float v = 0.f;
    if (K < Kin && N < Nout) {
        v = W[N * Kin + K];
        if (W2) v += W2[N * Kin + K];
    }
    dst[((size_t)frag << 9) + t] = (_Float16)v;
}

__global__ void vecadd_kernel(const float* __restrict__ a, const float* __restrict__ b,
                              float* __restrict__ o, int n)
{
    int i = blockIdx.x * blockDim.x + threadIdx.x;
    if (i < n) o[i] = a[i] + b[i];
}

// ---------------------------------------------------------------------------
// seq_mean + mean-removed transpose: x (B,1,F,S) -> xs (B,S,F), mean (B,F)
// ---------------------------------------------------------------------------
__global__ __launch_bounds__(192) void meanxs_kernel(const float* __restrict__ x,
                                                     float* __restrict__ mean,
                                                     float* __restrict__ xs)
{
    __shared__ float red[192];
    __shared__ float mval;
    int b = blockIdx.x / F_;
    int f = blockIdx.x % F_;
    int s = threadIdx.x;
    float v = (s < S_) ? x[(b * F_ + f) * S_ + s] : 0.f;
    red[s] = v;
    __syncthreads();
    for (int off = 96; off >= 6; off >>= 1) {
        if (s < off) red[s] += red[s + off];
        __syncthreads();
    }
    if (s == 0) {
        float t = 0.f;
        for (int i = 0; i < 6; ++i) t += red[i];
        mval = t * (1.0f / (float)S_);
        mean[b * F_ + f] = mval;
    }
    __syncthreads();
    if (s < S_) xs[(b * S_ + s) * F_ + f] = v - mval;
}

// ---------------------------------------------------------------------------
// Generic sum-over-periods SFF: out[b,s,c] = 0.25 * sum_p sff_p(in)[b,s,c]
// One block per (b,c) row.
// ---------------------------------------------------------------------------
__global__ __launch_bounds__(192) void sff_kernel(
    const float* __restrict__ in, int ld_in, int C,
    const float* __restrict__ c3,  const float* __restrict__ l3,
    const float* __restrict__ c6,  const float* __restrict__ l6,
    const float* __restrict__ c12, const float* __restrict__ l12,
    const float* __restrict__ c24, const float* __restrict__ l24,
    int wrow, float* __restrict__ out)
{
    __shared__ float xrow[S_];
    __shared__ float xc[S_];
    int b = blockIdx.x / C;
    int c = blockIdx.x % C;
    int s = threadIdx.x;
    if (s < S_) xrow[s] = in[(b * S_ + s) * ld_in + c];
    __syncthreads();

    const int periods[4] = {3, 6, 12, 24};
    const float* cw_all[4] = {c3, c6, c12, c24};
    const float* lw_all[4] = {l3, l6, l12, l24};
    float acc = 0.f;
    for (int pi = 0; pi < 4; ++pi) {
        int p = periods[pi];
        int pad = p >> 1;
        int kk = 2 * pad + 1;
        int seg = S_ / p;
        const float* cw = cw_all[pi] + wrow * kk;
        const float* lw = lw_all[pi] + wrow * seg * seg;
        if (s < S_) {
            float v = xrow[s];                       // residual add
            for (int j = 0; j < kk; ++j) {
                int idx = s + j - pad;
                float xv = (idx >= 0 && idx < S_) ? xrow[idx] : 0.f;
                v += cw[j] * xv;
            }
            xc[s] = v;
        }
        __syncthreads();
        if (s < S_) {
            int q = s / p, ph = s % p;
            const float* lrow = lw + q * seg;
            float sum = 0.f;
            for (int si = 0; si < seg; ++si) sum += lrow[si] * xc[si * p + ph];
            acc += sum;
        }
        __syncthreads();
    }
    if (s < S_) out[(b * S_ + s) * C + c] = 0.25f * acc;
}

// ---------------------------------------------------------------------------
// Persistent RNN scan: one WG (8 waves) per 16 batch rows, 168 steps.
// Weight residency: L0+L1 fragments in VGPRs (208/wave, loaded once);
// head + readout fragments in LDS (196 KB, copied once); activations in LDS.
// A fragments preloaded per layer so ds loads pipeline ahead of WMMA runs.
// ---------------------------------------------------------------------------
__global__ __launch_bounds__(256) void rnn_kernel(
    const float* __restrict__ x1,
    const _Float16* __restrict__ w0f,  const _Float16* __restrict__ w1f,
    const _Float16* __restrict__ wf1f, const _Float16* __restrict__ wf2f,
    const _Float16* __restrict__ wtabf, const _Float16* __restrict__ wfcf,
    const float* __restrict__ bb0,  const float* __restrict__ bb1,
    const float* __restrict__ bff1, const float* __restrict__ bff2,
    const float* __restrict__ btab, const float* __restrict__ bfc,
    float* __restrict__ ro)
{
    extern __shared__ __align__(128) char smem[];
    float*    hb  = (float*)(smem + LDS_HB);          // [3][16*128]
    _Float16* zin = (_Float16*)(smem + LDS_ZIN);      // [16*K0P]
    _Float16* z1s = (_Float16*)(smem + LDS_Z1);       // [16*U_]
    _Float16* z2s = (_Float16*)(smem + LDS_Z2);       // [16*U_]
    _Float16* hw0 = (_Float16*)(smem + LDS_HW);
    _Float16* hw1 = hw0 + NTH * KTH * 512;
    _Float16* hw2 = hw1 + NTH * KTH * 512;
    _Float16* wfl = hw2 + NTH * KTH * 512;

    const int tid  = threadIdx.x;
    const int wave = tid >> 5;
    const int lane = tid & 31;
    const int bbase = blockIdx.x * 16;
    const int nloc = lane & 15;
    const int mh   = (lane >> 4) << 3;
    const int nt0 = wave, nt1 = wave + 8;

    // one-time: head weights -> LDS
    copy_g2l(hw0, wf1f,  NTH * KTH * 512, tid, 256);
    copy_g2l(hw1, wf2f,  NTH * KTH * 512, tid, 256);
    copy_g2l(hw2, wtabf, NTH * KTH * 512, tid, 256);
    copy_g2l(wfl, wfcf,  KTC * 512,       tid, 256);

    // one-time: L0/L1 weights -> registers (this wave's n-tiles only)
    v16h w0r[2][KT0];
    v16h w1r[2][KT1];
#pragma unroll
    for (int kt = 0; kt < KT0; ++kt) {
        w0r[0][kt] = load_bfrag(w0f, nt0 * KT0 + kt, lane);
        w0r[1][kt] = load_bfrag(w0f, nt1 * KT0 + kt, lane);
    }
#pragma unroll
    for (int kt = 0; kt < KT1; ++kt) {
        w1r[0][kt] = load_bfrag(w1f, nt0 * KT1 + kt, lane);
        w1r[1][kt] = load_bfrag(w1f, nt1 * KT1 + kt, lane);
    }
    const float bias00 = bb0[nt0 * 16 + nloc];
    const float bias01 = bb0[nt1 * 16 + nloc];
    const float bias10 = bb1[nt0 * 16 + nloc];
    const float bias11 = bb1[nt1 * 16 + nloc];

    // loop-invariant head-job descriptors for this wave (3 jobs of 24)
    const _Float16* hwsel[3];
    float biash[3];
    int   hboff[3];     // base offset into hb for store
    int   istanh[3];
#pragma unroll
    for (int gi = 0; gi < 3; ++gi) {
        int g = wave + gi * 8;
        int head = g >> 3;
        int nt = g & 7;
        hwsel[gi]  = ((head == 0) ? hw0 : (head == 1) ? hw1 : hw2) + ((size_t)(nt * KTH) << 9);
        biash[gi]  = ((head == 0) ? bff1 : (head == 1) ? bff2 : btab)[nt * 16 + nloc];
        hboff[gi]  = head * (16 * H_) + nt * 16 + nloc;
        istanh[gi] = (head < 2);
    }

    for (int i = tid; i < 16 * K0P; i += 256) zin[i] = (_Float16)0.f;  // h0=0, pad=0
    __syncthreads();

    for (int t = 0; t < S_; ++t) {
        // stage xt -> zin[:, 0:12]
        if (tid < 16 * F_) {
            int m = tid / F_, c = tid % F_;
            zin[m * K0P + c] = (_Float16)x1[((bbase + m) * S_ + t) * F_ + c];
        }
        __syncthreads();

        // ---- Layer 0: z1 = silu(z0 @ Wb0^T + bb0)  (K=160 padded, N=256)
        {
            v16h a[KT0];
#pragma unroll
            for (int kt = 0; kt < KT0; ++kt) a[kt] = load_afrag(zin, K0P, lane, kt * 32);
            v8f acc0 = {}, acc1 = {};
#pragma unroll
            for (int kt = 0; kt < KT0; ++kt) {
                acc0 = WMMA_F16(a[kt], w0r[0][kt], acc0);
                acc1 = WMMA_F16(a[kt], w0r[1][kt], acc1);
            }
#pragma unroll
            for (int v = 0; v < 8; ++v) {
                z1s[(mh + v) * U_ + nt0 * 16 + nloc] = (_Float16)silu_f(acc0[v] + bias00);
                z1s[(mh + v) * U_ + nt1 * 16 + nloc] = (_Float16)silu_f(acc1[v] + bias01);
            }
        }
        __syncthreads();

        // ---- Layer 1: z2 = silu(z1 @ Wb1^T + bb1)  (K=256, N=256)
        {
            v16h a[KT1];
#pragma unroll
            for (int kt = 0; kt < KT1; ++kt) a[kt] = load_afrag(z1s, U_, lane, kt * 32);
            v8f acc0 = {}, acc1 = {};
#pragma unroll
            for (int kt = 0; kt < KT1; ++kt) {
                acc0 = WMMA_F16(a[kt], w1r[0][kt], acc0);
                acc1 = WMMA_F16(a[kt], w1r[1][kt], acc1);
            }
#pragma unroll
            for (int v = 0; v < 8; ++v) {
                z2s[(mh + v) * U_ + nt0 * 16 + nloc] = (_Float16)silu_f(acc0[v] + bias10);
                z2s[(mh + v) * U_ + nt1 * 16 + nloc] = (_Float16)silu_f(acc1[v] + bias11);
            }
        }
        __syncthreads();

        // ---- Heads: ff1 / ff2 / (ta+tb folded). 24 n-tiles over 8 waves;
        //      A fragments loaded once, reused for all 3 head jobs.
        {
            v16h a[KTH];
#pragma unroll
            for (int kt = 0; kt < KTH; ++kt) a[kt] = load_afrag(z2s, U_, lane, kt * 32);
#pragma unroll
            for (int gi = 0; gi < 3; ++gi) {
                const _Float16* wf = hwsel[gi];
                v8f acc = {};
#pragma unroll
                for (int kt = 0; kt < KTH; ++kt) {
                    v16h b = load_bfrag(wf, kt, lane);
                    acc = WMMA_F16(a[kt], b, acc);
                }
                float bv = biash[gi];
#pragma unroll
                for (int v = 0; v < 8; ++v) {
                    float xv = acc[v] + bv;
                    xv = istanh[gi] ? tanh_f(xv) : sigm_f(xv);
                    hb[hboff[gi] + (mh + v) * H_] = xv;
                }
            }
        }
        __syncthreads();

        // ---- h_new = ff1*(1-t) + t*ff2 -> zin[:, 12:140] (f16)
#pragma unroll
        for (int j = 0; j < 8; ++j) {
            int idx = tid + (j << 8);          // 0..2047
            int m = idx >> 7, n = idx & 127;
            float f1 = hb[idx];
            float f2 = hb[16 * H_ + idx];
            float tt = hb[2 * 16 * H_ + idx];
            zin[m * K0P + 12 + n] = (_Float16)(f1 * (1.f - tt) + tt * f2);
        }
        __syncthreads();

        // ---- readout (wave 0): r = h @ Wfc^T + bfc (O=3 of padded 16 cols)
        if (wave == 0) {
            v16h a[KTC];
#pragma unroll
            for (int kt = 0; kt < KTC; ++kt) a[kt] = load_afrag(zin, K0P, lane, 12 + kt * 32);
            v8f acc = {};
#pragma unroll
            for (int kt = 0; kt < KTC; ++kt) {
                v16h b = load_bfrag(wfl, kt, lane);
                acc = WMMA_F16(a[kt], b, acc);
            }
            if (nloc < O_) {
                float bv = bfc[nloc];
#pragma unroll
                for (int v = 0; v < 8; ++v)
                    ro[((bbase + mh + v) * S_ + t) * O_ + nloc] = acc[v] + bv;
            }
        }
        // next iteration's first barrier orders the xt restage
    }
}

// ---------------------------------------------------------------------------
// final: out[b,p,c] = sum_s (ro+x3)[b,s,c]*Wfits[c,p,s] + mean[b,c] + x2[b,p,c]
// ---------------------------------------------------------------------------
__global__ __launch_bounds__(192) void fits_kernel(
    const float* __restrict__ ro, const float* __restrict__ x3,
    const float* __restrict__ x2, const float* __restrict__ mean,
    const float* __restrict__ Wfits, float* __restrict__ out)
{
    __shared__ float ra[S_];
    int b = blockIdx.x / O_;
    int c = blockIdx.x % O_;
    int s = threadIdx.x;
    if (s < S_) {
        int idx = (b * S_ + s) * O_ + c;
        ra[s] = ro[idx] + x3[idx];
    }
    __syncthreads();
    if (s < S_) {
        const float* wrow = Wfits + (c * S_ + s) * S_;
        float o = 0.f;
        for (int k = 0; k < S_; ++k) o += ra[k] * wrow[k];
        out[(b * S_ + s) * O_ + c] = o + mean[b * F_ + c] + x2[(b * S_ + s) * O_ + c];
    }
}

extern "C" void kernel_launch(void* const* d_in, const int* in_sizes, int n_in,
                              void* d_out, int out_size, void* d_ws, size_t ws_size,
                              hipStream_t stream)
{
    const float* x    = (const float*)d_in[0];
    const float* c3   = (const float*)d_in[1];
    const float* l3   = (const float*)d_in[2];
    const float* c6   = (const float*)d_in[3];
    const float* l6   = (const float*)d_in[4];
    const float* c12  = (const float*)d_in[5];
    const float* l12  = (const float*)d_in[6];
    const float* c24  = (const float*)d_in[7];
    const float* l24  = (const float*)d_in[8];
    const float* Wb0  = (const float*)d_in[9];
    const float* bb0  = (const float*)d_in[10];
    const float* Wb1  = (const float*)d_in[11];
    const float* bb1  = (const float*)d_in[12];
    const float* Wff1 = (const float*)d_in[13];
    const float* bff1 = (const float*)d_in[14];
    const float* Wff2 = (const float*)d_in[15];
    const float* bff2 = (const float*)d_in[16];
    const float* Wta  = (const float*)d_in[17];
    const float* bta  = (const float*)d_in[18];
    const float* Wtb  = (const float*)d_in[19];
    const float* btb  = (const float*)d_in[20];
    const float* Wfc  = (const float*)d_in[21];
    const float* bfc  = (const float*)d_in[22];
    const float* Wfits= (const float*)d_in[23];
    float* out = (float*)d_out;
    (void)in_sizes; (void)n_in; (void)out_size; (void)ws_size;

    char* ws = (char*)d_ws;
    size_t off = 0;
    auto alloc = [&](size_t bytes) -> char* {
        char* p = ws + off;
        off = (off + bytes + 255) & ~(size_t)255;
        return p;
    };
    _Float16* w0f   = (_Float16*)alloc((size_t)KT0 * NT0 * 512 * 2);
    _Float16* w1f   = (_Float16*)alloc((size_t)KT1 * NT1 * 512 * 2);
    _Float16* wf1f  = (_Float16*)alloc((size_t)KTH * NTH * 512 * 2);
    _Float16* wf2f  = (_Float16*)alloc((size_t)KTH * NTH * 512 * 2);
    _Float16* wtabf = (_Float16*)alloc((size_t)KTH * NTH * 512 * 2);
    _Float16* wfcf  = (_Float16*)alloc((size_t)KTC * 1 * 512 * 2);
    float* btabw = (float*)alloc(H_ * sizeof(float));
    float* meanw = (float*)alloc((size_t)B_ * F_ * sizeof(float));
    float* xsw   = (float*)alloc((size_t)B_ * S_ * F_ * sizeof(float));
    float* x1w   = (float*)alloc((size_t)B_ * S_ * F_ * sizeof(float));
    float* x2w   = (float*)alloc((size_t)B_ * S_ * O_ * sizeof(float));
    float* x3w   = (float*)alloc((size_t)B_ * S_ * O_ * sizeof(float));
    float* row   = (float*)alloc((size_t)B_ * S_ * O_ * sizeof(float));

    // weight prep
    swizzle_kernel<<<KT0 * NT0, 512, 0, stream>>>(Wb0, nullptr, w0f, U_, 140, KT0);
    swizzle_kernel<<<KT1 * NT1, 512, 0, stream>>>(Wb1, nullptr, w1f, U_, U_, KT1);
    swizzle_kernel<<<KTH * NTH, 512, 0, stream>>>(Wff1, nullptr, wf1f, H_, U_, KTH);
    swizzle_kernel<<<KTH * NTH, 512, 0, stream>>>(Wff2, nullptr, wf2f, H_, U_, KTH);
    swizzle_kernel<<<KTH * NTH, 512, 0, stream>>>(Wta, Wtb, wtabf, H_, U_, KTH);
    swizzle_kernel<<<KTC, 512, 0, stream>>>(Wfc, nullptr, wfcf, O_, H_, KTC);
    vecadd_kernel<<<1, 128, 0, stream>>>(bta, btb, btabw, H_);

    // preprocessing
    meanxs_kernel<<<B_ * F_, 192, 0, stream>>>(x, meanw, xsw);
    sff_kernel<<<B_ * F_, 192, 0, stream>>>(xsw, F_, F_, c3, l3, c6, l6, c12, l12, c24, l24, 0, x1w);
    sff_kernel<<<B_ * O_, 192, 0, stream>>>(x1w, F_, O_, c3, l3, c6, l6, c12, l12, c24, l24, 1, x2w);
    sff_kernel<<<B_ * O_, 192, 0, stream>>>(x1w, F_, O_, c3, l3, c6, l6, c12, l12, c24, l24, 2, x3w);

    // recurrent scan (dominant cost, WMMA; 241 KB dynamic LDS per WG)
    rnn_kernel<<<B_ / 16, 256, LDS_TOTAL, stream>>>(x1w, w0f, w1f, wf1f, wf2f, wtabf, wfcf,
                                                    bb0, bb1, bff1, bff2, btabw, bfc, row);

    // fits projection + residuals
    fits_kernel<<<B_ * O_, 192, 0, stream>>>(row, x3w, x2w, meanw, Wfits, out);
}
